// FullSelfAttention_32341103739031
// MI455X (gfx1250) — matrix operand order repaired and checked by reading, not compiled, not run on hardware
//
#include <hip/hip_runtime.h>

// Problem constants (match reference)
#define Hh   16
#define Dd   128
#define Tn   2048
#define Bn   4
#define Ee   2048
#define Mrows (Bn * Tn)   // 8192

typedef unsigned short u16;
typedef __attribute__((ext_vector_type(16))) __bf16 v16bf;
typedef __attribute__((ext_vector_type(8)))  __bf16 v8bf;
typedef __attribute__((ext_vector_type(8)))  float  v8f;
typedef int v4i_ __attribute__((vector_size(16)));   // matches builtin's pointee type

#define AS1 __attribute__((address_space(1)))
#define AS3 __attribute__((address_space(3)))

#if __has_builtin(__builtin_amdgcn_global_load_async_to_lds_b128) && \
    __has_builtin(__builtin_amdgcn_s_wait_asynccnt)
#define HAVE_ASYNC 1
#else
#define HAVE_ASYNC 0
#endif

// ---------------------------------------------------------------- helpers

__device__ __forceinline__ u16 f2bf(float f) {
    union { float f; unsigned u; } x; x.f = f;
    unsigned u = x.u;
    unsigned r = (u + 0x7FFFu + ((u >> 16) & 1u)) >> 16;   // round-nearest-even
    return (u16)r;
}

// A fragment (16x32 bf16, MxK) from row-major memory, base -> A[row0][k0].
// wave32 layout: lane<16: M=lane, K elems 0..7,16..23 ; lane>=16: M=lane-16, K 8..15,24..31
__device__ __forceinline__ v16bf load_a_frag(const u16* base, int ld, int lane) {
    int m  = lane & 15;
    int kb = (lane >> 4) << 3;            // 0 or 8
    const u16* p = base + (size_t)m * ld;
    v8bf lo = *(const v8bf*)(p + kb);
    v8bf hi = *(const v8bf*)(p + 16 + kb);
    v16bf a;
#pragma unroll
    for (int i = 0; i < 8; ++i) { a[i] = lo[i]; a[i + 8] = hi[i]; }
    return a;
}

// B fragment (32x16 bf16 KxN) from BT[n][k] row-major, base -> BT[col0][k0] (32B-aligned rows).
__device__ __forceinline__ v16bf load_b_frag(const u16* base, int ld, int lane) {
    int n   = lane & 15;
    int kb2 = (lane >> 4) << 3;           // 0 or 8
    const u16* p = base + (size_t)n * ld;
    v8bf lo = *(const v8bf*)(p + kb2);
    v8bf hi = *(const v8bf*)(p + 16 + kb2);
    v16bf b;
#pragma unroll
    for (int i = 0; i < 8; ++i) { b[i] = lo[i]; b[i + 8] = hi[i]; }
    return b;
}

__device__ __forceinline__ v8f wmma_bf16(v16bf a, v16bf b, v8f c) {
    return __builtin_amdgcn_wmma_f32_16x16x32_bf16(false, a, false, b, (short)0, c,
                                                   false, false);
}

__device__ __forceinline__ void wait_async_pending() {   // current tile landed, next in flight
#if HAVE_ASYNC
    __builtin_amdgcn_s_wait_asynccnt(4);
#else
    asm volatile("s_wait_dscnt 0x0" ::: "memory");
#endif
}
__device__ __forceinline__ void wait_async_zero() {
#if HAVE_ASYNC
    __builtin_amdgcn_s_wait_asynccnt(0);
#else
    asm volatile("s_wait_dscnt 0x0" ::: "memory");
#endif
}

// ---------------------------------------------------------------- converts

__global__ __launch_bounds__(256) void cvt_bf16_kernel(const float* __restrict__ in,
                                                       u16* __restrict__ out, long n) {
    long i = (long)blockIdx.x * blockDim.x + threadIdx.x;
    if (i < n) out[i] = f2bf(in[i]);
}

// w: [E][E] f32 (k rows, n cols)  ->  wt: [E][E] bf16 transposed (n rows, k cols)
__global__ __launch_bounds__(256) void wt_transpose_kernel(const float* __restrict__ w,
                                                           u16* __restrict__ wt) {
    long idx = (long)blockIdx.x * blockDim.x + threadIdx.x;
    int k = (int)(idx / Ee), n = (int)(idx % Ee);
    wt[(long)n * Ee + k] = f2bf(w[idx]);
}

// ---------------------------------------------------------------- GEMM
// C(f32 MxN) = A(bf16 MxK row-major) * BT(bf16 NxK row-major)^T
// Block: 256 threads (8 waves, 2Mx4N), tile 128x128, BK=32, LDS double-buffered,
// staged with GLOBAL_LOAD_ASYNC_TO_LDS_B128 (ASYNCcnt) when available.

#define LDPAD 40   // u16 per LDS row: 32 data + 8 pad -> 80B stride, conflict-free b128 reads

__global__ __launch_bounds__(256) void gemm_bf16_f32(const u16* __restrict__ A,
                                                     const u16* __restrict__ BT,
                                                     float* __restrict__ C,
                                                     int M, int N, int K) {
    __shared__ __attribute__((aligned(16))) u16 lA[2][128][LDPAD];
    __shared__ __attribute__((aligned(16))) u16 lB[2][128][LDPAD];

    int tid  = threadIdx.x;
    int lane = tid & 31;
    int w    = tid >> 5;
    int wm   = w >> 2;          // 0..1  (64-row slice)
    int wn   = w & 3;           // 0..3  (32-col slice)

    int tilesN = N >> 7;
    int row0 = (blockIdx.x / tilesN) << 7;
    int col0 = (blockIdx.x % tilesN) << 7;

    const u16* gA = A  + (size_t)row0 * K;
    const u16* gB = BT + (size_t)col0 * K;

    // stage one BK=32 tile (A:128x32, B:128x32) into LDS buffer `buf`
    auto stage = [&](int k0, int buf) {
#pragma unroll
        for (int half = 0; half < 2; ++half) {
            int c   = tid + half * 256;      // 512 chunks of 16B per matrix
            int row = c >> 2;
            int kc  = (c & 3) << 3;          // u16 offset within BK
#if HAVE_ASYNC
            __builtin_amdgcn_global_load_async_to_lds_b128(
                (AS1 v4i_*)(gA + (size_t)row * K + k0 + kc),
                (AS3 v4i_*)&lA[buf][row][kc], 0, 0);
            __builtin_amdgcn_global_load_async_to_lds_b128(
                (AS1 v4i_*)(gB + (size_t)row * K + k0 + kc),
                (AS3 v4i_*)&lB[buf][row][kc], 0, 0);
#else
            uint4 va = *(const uint4*)(gA + (size_t)row * K + k0 + kc);
            uint4 vb = *(const uint4*)(gB + (size_t)row * K + k0 + kc);
            *(uint4*)&lA[buf][row][kc] = va;
            *(uint4*)&lB[buf][row][kc] = vb;
#endif
        }
    };

    const v8f vzero = {};
    v8f acc[4][2];
#pragma unroll
    for (int i = 0; i < 4; ++i)
#pragma unroll
        for (int j = 0; j < 2; ++j) acc[i][j] = vzero;

    int m   = lane & 15;
    int kb  = (lane >> 4) << 3;   // A frag k-half (u16)
    int kb2 = (lane >> 4) << 4;   // unused; B frag reads use kb

    stage(0, 0);

    int nk = K >> 5;
    for (int kt = 0; kt < nk; ++kt) {
        int cur = kt & 1;
        if (kt + 1 < nk) {
            stage((kt + 1) << 5, cur ^ 1);   // 4 async b128 per thread in flight
            wait_async_pending();            // in-order completion: current tile landed
        } else {
            wait_async_zero();
        }
        __syncthreads();

        // fragments from LDS
        v16bf af[4];
#pragma unroll
        for (int i = 0; i < 4; ++i) {
            const u16* p = &lA[cur][wm * 64 + 16 * i + m][0];
            v8bf lo = *(const v8bf*)(p + kb);
            v8bf hi = *(const v8bf*)(p + 16 + kb);
#pragma unroll
            for (int e = 0; e < 8; ++e) { af[i][e] = lo[e]; af[i][e + 8] = hi[e]; }
        }
        v16bf bfr[2];
#pragma unroll
        for (int j = 0; j < 2; ++j) {
            const u16* p = &lB[cur][wn * 32 + 16 * j + m][0] + kb;
            v8bf lo = *(const v8bf*)p;
            v8bf hi = *(const v8bf*)(p + 16);
#pragma unroll
            for (int e = 0; e < 8; ++e) { bfr[j][e] = lo[e]; bfr[j][e + 8] = hi[e]; }
        }
#pragma unroll
        for (int i = 0; i < 4; ++i)
#pragma unroll
            for (int j = 0; j < 2; ++j)
                acc[i][j] = wmma_bf16(af[i], bfr[j], acc[i][j]);

        __syncthreads();   // all waves done reading `cur` before it is restaged
    }

    // epilogue: scalar base + 32-bit lane offsets
    float* Cw = C + (size_t)(row0 + wm * 64) * N + col0 + wn * 32;
    int n  = lane & 15;
    int mb = (lane >> 4) << 3;
#pragma unroll
    for (int i = 0; i < 4; ++i)
#pragma unroll
        for (int j = 0; j < 2; ++j)
#pragma unroll
            for (int r = 0; r < 8; ++r)
                Cw[(16 * i + mb + r) * N + 16 * j + n] = acc[i][j][r];
}

// ---------------------------------------------------------------- RoPE + RMSNorm
// in: f32 [B,T,H,D] ; out: bf16 [B,H,T,D]

__global__ __launch_bounds__(128) void rope_rms_kernel(const float* __restrict__ in,
                                                       u16* __restrict__ out) {
    __shared__ float red[4];
    int i = blockIdx.x;                 // (b*T + t)*H + h
    int h = i % Hh;
    int t = (i / Hh) % Tn;
    int b = i / (Hh * Tn);
    int d = threadIdx.x;

    const float* p = in + (size_t)i * Dd;
    int dm = d & 63;
    float invf = __expf(-(float)dm * 0.14391157f);   // ln(10000)/64
    float th = (float)t * invf;
    float c = __cosf(th), s = __sinf(th);
    float x1 = p[dm], x2 = p[dm + 64];
    float y = (d < 64) ? (x1 * c - x2 * s) : (x1 * s + x2 * c);

    float sq = y * y;
#pragma unroll
    for (int off = 1; off < 32; off <<= 1) sq += __shfl_xor(sq, off, 32);
    if ((threadIdx.x & 31) == 0) red[threadIdx.x >> 5] = sq;
    __syncthreads();
    float ms = (red[0] + red[1] + red[2] + red[3]) * (1.0f / 128.0f);
    float g = rsqrtf(ms + 1.1920929e-07f);

    out[(((size_t)b * Hh + h) * Tn + t) * Dd + d] = f2bf(y * g);
}

// in: f32 [B,T,H,D] ; out: bf16 [B,H,D,T]
__global__ __launch_bounds__(128) void v_transpose_kernel(const float* __restrict__ in,
                                                          u16* __restrict__ out) {
    int i = blockIdx.x;                 // (b*T + t)*H + h
    int h = i % Hh;
    int t = (i / Hh) % Tn;
    int b = i / (Hh * Tn);
    int d = threadIdx.x;
    out[(((size_t)b * Hh + h) * Dd + d) * Tn + t] = f2bf(in[(size_t)i * Dd + d]);
}

// ---------------------------------------------------------------- Flash attention
// Qb,Kb: bf16 [B,H,T,D] ; VbT: bf16 [B,H,D,T] ; Yb out: bf16 [B,T,E]

__global__ __launch_bounds__(128) void flash_attn_kernel(const u16* __restrict__ Qb,
                                                         const u16* __restrict__ Kb,
                                                         const u16* __restrict__ VbT,
                                                         u16* __restrict__ Yb) {
    __shared__ __attribute__((aligned(16))) u16 psh[4][16][32];
    int lane = threadIdx.x & 31;
    int w    = threadIdx.x >> 5;
    int gw   = blockIdx.x * 4 + w;      // (b, h, qtile)
    int qt = gw & 127;
    int h  = (gw >> 7) & (Hh - 1);
    int b  = gw >> 11;
    int q0 = qt << 4;

    const u16* Qh = Qb  + ((size_t)b * Hh + h) * Tn * Dd;
    const u16* Kh = Kb  + ((size_t)b * Hh + h) * Tn * Dd;
    const u16* Vh = VbT + ((size_t)b * Hh + h) * Dd * Tn;

    v16bf qf[4];
#pragma unroll
    for (int c = 0; c < 4; ++c)
        qf[c] = load_a_frag(Qh + (size_t)q0 * Dd + 32 * c, Dd, lane);

    const v8f vzero = {};
    v8f ot[8];
    float mrow[8], lrow[8];
#pragma unroll
    for (int dt = 0; dt < 8; ++dt) ot[dt] = vzero;
#pragma unroll
    for (int r = 0; r < 8; ++r) { mrow[r] = -1e30f; lrow[r] = 0.0f; }

    int n  = lane & 15;
    int mb = (lane >> 4) << 3;
    const float scale = 0.08838834764831845f;   // 1/sqrt(128)
    int nblocks = (q0 + 15) / 32 + 1;

    for (int kblk = 0; kblk < nblocks; ++kblk) {
        int k0 = kblk * 32;

        // S = Q * K^T  (two 16x16 tiles over keys k0..k0+31)
        v8f s[2]; s[0] = vzero; s[1] = vzero;
#pragma unroll
        for (int t2 = 0; t2 < 2; ++t2)
#pragma unroll
            for (int c = 0; c < 4; ++c) {
                v16bf kf = load_b_frag(Kh + (size_t)(k0 + 16 * t2) * Dd + 32 * c, Dd, lane);
                s[t2] = wmma_bf16(qf[c], kf, s[t2]);
            }

        // causal mask + online softmax
#pragma unroll
        for (int r = 0; r < 8; ++r) {
            int qrow = q0 + mb + r;
            float s0 = (k0 + n      <= qrow) ? s[0][r] * scale : -1e30f;
            float s1 = (k0 + 16 + n <= qrow) ? s[1][r] * scale : -1e30f;
            float mx = fmaxf(s0, s1);
#pragma unroll
            for (int off = 1; off < 16; off <<= 1) mx = fmaxf(mx, __shfl_xor(mx, off, 32));
            float mnew = fmaxf(mrow[r], mx);
            float corr = __expf(mrow[r] - mnew);
            float p0 = __expf(s0 - mnew), p1 = __expf(s1 - mnew);
            float ps = p0 + p1;
#pragma unroll
            for (int off = 1; off < 16; off <<= 1) ps += __shfl_xor(ps, off, 32);
            lrow[r] = lrow[r] * corr + ps;
            mrow[r] = mnew;
#pragma unroll
            for (int dt = 0; dt < 8; ++dt) ot[dt][r] *= corr;
            psh[w][mb + r][n]      = f2bf(p0);
            psh[w][mb + r][16 + n] = f2bf(p1);
        }

        // same-wave LDS ordering before the cross-lane transpose read
        asm volatile("s_wait_dscnt 0x0" ::: "memory");

        int m  = lane & 15;
        int ko = (lane >> 4) << 3;
        v8bf plo = *(const v8bf*)&psh[w][m][ko];
        v8bf phi = *(const v8bf*)&psh[w][m][16 + ko];
        v16bf pf;
#pragma unroll
        for (int i = 0; i < 8; ++i) { pf[i] = plo[i]; pf[i + 8] = phi[i]; }

        // O += P * V
#pragma unroll
        for (int dt = 0; dt < 8; ++dt) {
            v16bf vf = load_b_frag(Vh + (size_t)(dt * 16) * Tn + k0, Tn, lane);
            ot[dt] = wmma_bf16(pf, vf, ot[dt]);
        }
    }

    // normalize and store bf16 rows into Yb [B,T,E]
    u16* Yrow = Yb + ((size_t)b * Tn + q0) * Ee + h * Dd;
#pragma unroll
    for (int r = 0; r < 8; ++r) {
        float inv = 1.0f / lrow[r];
#pragma unroll
        for (int dt = 0; dt < 8; ++dt)
            Yrow[(mb + r) * Ee + dt * 16 + n] = f2bf(ot[dt][r] * inv);
    }
}

// ---------------------------------------------------------------- launch

extern "C" void kernel_launch(void* const* d_in, const int* in_sizes, int n_in,
                              void* d_out, int out_size, void* d_ws, size_t ws_size,
                              hipStream_t stream) {
    (void)in_sizes; (void)n_in; (void)out_size; (void)ws_size;
    const float* x  = (const float*)d_in[0];
    const float* wq = (const float*)d_in[1];
    const float* wk = (const float*)d_in[2];
    const float* wv = (const float*)d_in[3];
    const float* wo = (const float*)d_in[4];
    float* out = (float*)d_out;

    char* ws = (char*)d_ws;
    size_t off = 0;
    const size_t xbBytes  = (size_t)Mrows * Ee * sizeof(u16);
    const size_t wBytes   = (size_t)Ee * Ee * sizeof(u16);
    const size_t tmpBytes = (size_t)Mrows * Ee * sizeof(float);

    u16* xb   = (u16*)(ws + off); off += xbBytes;
    u16* wqT  = (u16*)(ws + off); off += wBytes;
    u16* wkT  = (u16*)(ws + off); off += wBytes;
    u16* wvT  = (u16*)(ws + off); off += wBytes;
    u16* woT  = (u16*)(ws + off); off += wBytes;
    float* tmpF = (float*)(ws + off); off += tmpBytes;
    u16* Qb   = (u16*)(ws + off); off += xbBytes;
    u16* Kb   = (u16*)(ws + off); off += xbBytes;
    u16* VbT  = (u16*)(ws + off); off += xbBytes;
    u16* Yb   = (u16*)(ws + off); off += xbBytes;

    const long nx = (long)Mrows * Ee;
    cvt_bf16_kernel<<<(nx + 255) / 256, 256, 0, stream>>>(x, xb, nx);
    const int wtBlocks = (Ee * Ee) / 256;
    wt_transpose_kernel<<<wtBlocks, 256, 0, stream>>>(wq, wqT);
    wt_transpose_kernel<<<wtBlocks, 256, 0, stream>>>(wk, wkT);
    wt_transpose_kernel<<<wtBlocks, 256, 0, stream>>>(wv, wvT);
    wt_transpose_kernel<<<wtBlocks, 256, 0, stream>>>(wo, woT);

    const int gemmBlocks = (Mrows / 128) * (Ee / 128);   // 128x128 tile / block
    const int headBlocks = Mrows * Hh;                   // one block per (b,t,h)

    gemm_bf16_f32<<<gemmBlocks, 256, 0, stream>>>(xb, wqT, tmpF, Mrows, Ee, Ee);
    rope_rms_kernel<<<headBlocks, 128, 0, stream>>>(tmpF, Qb);
    gemm_bf16_f32<<<gemmBlocks, 256, 0, stream>>>(xb, wkT, tmpF, Mrows, Ee, Ee);
    rope_rms_kernel<<<headBlocks, 128, 0, stream>>>(tmpF, Kb);
    gemm_bf16_f32<<<gemmBlocks, 256, 0, stream>>>(xb, wvT, tmpF, Mrows, Ee, Ee);
    v_transpose_kernel<<<headBlocks, 128, 0, stream>>>(tmpF, VbT);

    const int flashBlocks = (Bn * Hh * (Tn / 16)) / 4;   // 4 waves / block
    flash_attn_kernel<<<flashBlocks, 128, 0, stream>>>(Qb, Kb, VbT, Yb);

    gemm_bf16_f32<<<gemmBlocks, 256, 0, stream>>>(Yb, woT, out, Mrows, Ee, Ee);
}
